// SingleHeadCrossAttention_12481174962389
// MI455X (gfx1250) — matrix operand order repaired and checked by reading, not compiled, not run on hardware
//
#include <hip/hip_runtime.h>
#include <math.h>

// Problem constants (match reference)
#define B_ 8
#define T_ 2048
#define E_ 1024
#define H_ 512

typedef __bf16 bf16;
typedef __attribute__((ext_vector_type(16))) __bf16 v16bf;
typedef __attribute__((ext_vector_type(8)))  float v8f;
typedef __attribute__((ext_vector_type(4)))  unsigned int v4u;
typedef __attribute__((ext_vector_type(8)))  int v8i;
typedef __attribute__((ext_vector_type(4)))  int v4i;

union FragBF {
    v16bf v;
    uint4 q[2];
};

__device__ inline v8f zero8() {
    v8f z = {0.f, 0.f, 0.f, 0.f, 0.f, 0.f, 0.f, 0.f};
    return z;
}

// ---------------------------------------------------------------------------
// Tensor Data Mover: 2D tile load Global -> LDS.
// D# packing per CDNA5 ISA:
//   group0: [1:0]=count=1, [63:32]=lds_addr, [120:64]=global_addr, [127:126]=type=2
//   group1: [17:16]=data_size, [79:48]=tensor_dim0, [111:80]=tensor_dim1,
//           [127:112]=tile_dim0, [143:128]=tile_dim1, [207:160]=tensor_dim0_stride
// data_size_code: 0=1B,1=2B,2=4B,3=8B; dims/strides in data_size units.
// This toolchain's builtin is the 6-arg form:
//   (uint32x4 g0, int32x8 g1, int32x4 g2, int32x4 g3, int32x8 extra, i32 cpol)
// ---------------------------------------------------------------------------
__device__ inline void tdm_load_2d_to_lds(void* lds_dst, const void* gsrc,
                                          unsigned tile_d0, unsigned tile_d1,
                                          unsigned long long stride_d0_units,
                                          unsigned data_size_code)
{
    const unsigned long long gaddr = (unsigned long long)(uintptr_t)gsrc;
    const unsigned laddr = (unsigned)(uintptr_t)lds_dst;   // LDS byte offset
    const unsigned long long tdim0 = 0x400000ull;          // generous bounds: no OOB zeroing
    const unsigned long long tdim1 = 0x400000ull;

    v4u g0;
    g0[0] = 1u;                                            // count=1, user descriptor
    g0[1] = laddr;                                         // lds_addr
    g0[2] = (unsigned)(gaddr & 0xffffffffu);               // global_addr[31:0]
    g0[3] = (unsigned)((gaddr >> 32) & 0x1ffffffu)         // global_addr[56:32]
            | (2u << 30);                                  // type = 2 ("image")

    v8i g1;
    g1[0] = (int)(data_size_code << 16);                   // wg_mask=0, data_size
    g1[1] = (int)((tdim0 & 0xffffu) << 16);                // tensor_dim0[15:0] @63:48
    g1[2] = (int)(((tdim0 >> 16) & 0xffffu)                // tensor_dim0[31:16]
            | ((tdim1 & 0xffffu) << 16));                  // tensor_dim1[15:0]
    g1[3] = (int)(((tdim1 >> 16) & 0xffffu)                // tensor_dim1[31:16]
            | ((tile_d0 & 0xffffu) << 16));                // tile_dim0
    g1[4] = (int)(tile_d1 & 0xffffu);                      // tile_dim1; tile_dim2=0
    g1[5] = (int)(stride_d0_units & 0xffffffffu);          // tensor_dim0_stride[31:0]
    g1[6] = (int)((stride_d0_units >> 32) & 0xffffu);      // stride[47:32]; dim1_stride=0
    g1[7] = 0;

    v4i z4 = {0, 0, 0, 0};
    v8i z8 = {0, 0, 0, 0, 0, 0, 0, 0};
    __builtin_amdgcn_tensor_load_to_lds(g0, g1, z4, z4, z8, 0);
}

// ---------------------------------------------------------------------------
// Kernel 1: C(M,N) = X(M,K) * W(K,N), f32 inputs -> bf16 output
// M = B*T = 16384, K = E = 1024, N = H = 512
// Block tile 128x64, k-chunk 32, 8 waves (4 along M x 2 along N),
// each wave computes 2x2 tiles of 16x16 via v_wmma_f32_16x16x32_bf16.
// ---------------------------------------------------------------------------
__global__ __launch_bounds__(256)
void qkv_gemm_kernel(const float* __restrict__ X, const float* __restrict__ W,
                     bf16* __restrict__ Out)
{
    constexpr int K  = E_;
    constexpr int N  = H_;
    constexpr int BM = 128, BN = 64, KC = 32;

    __shared__ bf16 As[BM * KC];   // (m,k) row-major, stride KC
    __shared__ bf16 Bst[BN * KC];  // transposed: (n,k), stride KC

    const int m0   = blockIdx.x * BM;
    const int n0   = blockIdx.y * BN;
    const int tid  = threadIdx.x;
    const int wave = tid >> 5;
    const int lane = tid & 31;
    const int wm   = wave & 3;   // wave position along M (0..3)
    const int wn   = wave >> 2;  // wave position along N (0..1)
    const int l16  = lane & 15;
    const int lh   = lane >> 4;  // lane half (0/1)

    v8f acc[2][2];
    acc[0][0] = zero8(); acc[0][1] = zero8();
    acc[1][0] = zero8(); acc[1][1] = zero8();

    for (int k0 = 0; k0 < K; k0 += KC) {
        __syncthreads();
        // A tile: 128x32 f32 -> bf16 (1024 float4, 4 per thread)
        for (int g = 0; g < 4; ++g) {
            int f  = tid + g * 256;       // float4 index; 8 float4 per row
            int r  = f >> 3;
            int c4 = f & 7;
            float4 xv = *(const float4*)(X + (size_t)(m0 + r) * K + k0 + c4 * 4);
            bf16* dst = &As[r * KC + c4 * 4];
            dst[0] = (bf16)xv.x; dst[1] = (bf16)xv.y;
            dst[2] = (bf16)xv.z; dst[3] = (bf16)xv.w;
        }
        // B tile: 32x64 f32 from W(k,n), stored transposed (n,k) as bf16
        for (int g = 0; g < 2; ++g) {
            int f  = tid + g * 256;       // float4 index over 32*16
            int kk = f >> 4;
            int c4 = f & 15;
            float4 wv = *(const float4*)(W + (size_t)(k0 + kk) * N + n0 + c4 * 4);
            Bst[(c4 * 4 + 0) * KC + kk] = (bf16)wv.x;
            Bst[(c4 * 4 + 1) * KC + kk] = (bf16)wv.y;
            Bst[(c4 * 4 + 2) * KC + kk] = (bf16)wv.z;
            Bst[(c4 * 4 + 3) * KC + kk] = (bf16)wv.w;
        }
        // Prefetch next k-slice into L2 (global_prefetch_b8)
        if (k0 + KC < K) {
            __builtin_prefetch((const void*)(X + (size_t)(m0 + (tid & 127)) * K + (k0 + KC)), 0, 0);
            __builtin_prefetch((const void*)(W + (size_t)(k0 + KC + (tid & 31)) * N + n0), 0, 0);
        }
        __syncthreads();

        FragBF a[2], b[2];
        for (int mt = 0; mt < 2; ++mt) {
            const bf16* ar = &As[(wm * 32 + mt * 16 + l16) * KC];
            a[mt].q[0] = *(const uint4*)(ar + lh * 8);        // K 0..7  / 8..15
            a[mt].q[1] = *(const uint4*)(ar + 16 + lh * 8);   // K 16..23 / 24..31
        }
        for (int nt = 0; nt < 2; ++nt) {
            const bf16* br = &Bst[(wn * 32 + nt * 16 + l16) * KC];
            b[nt].q[0] = *(const uint4*)(br + lh * 8);        // K 0..7  / 8..15
            b[nt].q[1] = *(const uint4*)(br + 16 + lh * 8);   // K 16..23 / 24..31
        }
        for (int mt = 0; mt < 2; ++mt)
            for (int nt = 0; nt < 2; ++nt)
                acc[mt][nt] = __builtin_amdgcn_wmma_f32_16x16x32_bf16(
                    false, a[mt].v, false, b[nt].v,
                    (short)0, acc[mt][nt], false, false);
    }

    // Epilogue: f32 accum -> bf16 store
    for (int mt = 0; mt < 2; ++mt)
        for (int nt = 0; nt < 2; ++nt)
            for (int vv = 0; vv < 8; ++vv) {
                int row = m0 + wm * 32 + mt * 16 + vv + lh * 8;
                int col = n0 + wn * 32 + nt * 16 + l16;
                Out[(size_t)row * N + col] = (bf16)acc[mt][nt][vv];
            }
}

// ---------------------------------------------------------------------------
// Kernel 2: flash attention over bf16 q,k,v -> f32 out
// One block per (batch, 64-row q block). 256 threads = 8 waves.
// Q block and K chunks staged via the Tensor Data Mover (wave 0 issues,
// s_wait_tensorcnt + barrier before consumption). S = Q K^T computed
// cooperatively (2 tiles of 16x16 per wave), online softmax rows in LDS,
// then each wave owns a 64-wide slice of H for P*V.
// ---------------------------------------------------------------------------
constexpr int BQ  = 64;
constexpr int BKV = 64;
constexpr int HC  = 64;

constexpr int OFF_Q  = 0;                         // 64x512 bf16
constexpr int OFF_K  = OFF_Q + BQ * H_ * 2;       // 64x64 bf16 (kv,h)
constexpr int OFF_VT = OFF_K + BKV * HC * 2;      // 8 waves x (64h x 64kv) bf16
constexpr int OFF_S  = OFF_VT + 8 * HC * BKV * 2; // 64x64 f32
constexpr int OFF_P  = OFF_S + BQ * BKV * 4;      // 64x64 bf16
constexpr int OFF_M  = OFF_P + BQ * BKV * 2;
constexpr int OFF_L  = OFF_M + 256;
constexpr int OFF_C  = OFF_L + 256;
constexpr int ATTN_LDS_BYTES = OFF_C + 256;       // 164608

__global__ __launch_bounds__(256)
void flash_attn_kernel(const bf16* __restrict__ Qg, const bf16* __restrict__ Kg,
                       const bf16* __restrict__ Vg, float* __restrict__ Out,
                       const int* __restrict__ should_mask)
{
    extern __shared__ char smem[];
    bf16*  Qs   = (bf16*)(smem + OFF_Q);
    bf16*  Kc   = (bf16*)(smem + OFF_K);
    bf16*  Vt   = (bf16*)(smem + OFF_VT);
    float* Sc   = (float*)(smem + OFF_S);
    bf16*  Ps   = (bf16*)(smem + OFF_P);
    float* mrow = (float*)(smem + OFF_M);
    float* lrow = (float*)(smem + OFF_L);
    float* crow = (float*)(smem + OFF_C);

    const float scale = 22.627416997969522f;  // sqrt(512): faithful to reference

    const int qb    = blockIdx.x;     // 0..31
    const int bat   = blockIdx.y;     // 0..7
    const int qBase = qb * BQ;
    const int tid   = threadIdx.x;
    const int wave  = tid >> 5;
    const int lane  = tid & 31;
    const int l16   = lane & 15;
    const int lh    = lane >> 4;

    const size_t baseQ = ((size_t)bat * T_ + qBase) * H_;

    // TDM: stage whole Q block (64 rows x 1024B, contiguous) as one 2D tile
    if (wave == 0) {
        tdm_load_2d_to_lds(Qs, Qg + baseQ,
                           /*tile_d0=*/(H_ * 2) / 8, /*tile_d1=*/BQ,
                           /*stride_d0=*/(H_ * 2) / 8, /*data_size=8B*/3);
        __builtin_amdgcn_s_wait_tensorcnt(0);
    }
    if (tid < BQ) { mrow[tid] = -INFINITY; lrow[tid] = 0.f; }
    __syncthreads();

    v8f oacc[4][4];
    for (int mt = 0; mt < 4; ++mt)
        for (int nt = 0; nt < 4; ++nt)
            oacc[mt][nt] = zero8();

    const int mask = *should_mask;
    const int jmax = mask ? qb : (T_ / BKV - 1);
    const int tile0 = wave * 2;  // this wave's 2 score tiles (of 16)

    for (int j = 0; j <= jmax; ++j) {
        const size_t baseKV = ((size_t)bat * T_ + j * BKV) * H_;

        v8f sacc[2];
        sacc[0] = zero8(); sacc[1] = zero8();

        // ---- S = Q K^T over H in 64-wide chunks ----
        for (int hc = 0; hc < H_ / HC; ++hc) {
            __syncthreads();
            // TDM: K chunk 64 kv rows x 128B, global row stride 1024B
            if (wave == 0) {
                tdm_load_2d_to_lds(Kc, Kg + baseKV + hc * HC,
                                   /*tile_d0=*/(HC * 2) / 8, /*tile_d1=*/BKV,
                                   /*stride_d0=*/(H_ * 2) / 8, /*data_size=8B*/3);
                __builtin_amdgcn_s_wait_tensorcnt(0);
            }
            __syncthreads();

            for (int t = 0; t < 2; ++t) {
                const int tile = tile0 + t;
                const int mt = tile >> 2, nt = tile & 3;
                for (int ks = 0; ks < 2; ++ks) {
                    FragBF a, bfr;
                    const bf16* ar = &Qs[(mt * 16 + l16) * H_ + hc * HC + ks * 32];
                    a.q[0] = *(const uint4*)(ar + lh * 8);
                    a.q[1] = *(const uint4*)(ar + 16 + lh * 8);
                    const bf16* br = &Kc[(nt * 16 + l16) * HC + ks * 32];
                    bfr.q[0] = *(const uint4*)(br + lh * 8);
                    bfr.q[1] = *(const uint4*)(br + 16 + lh * 8);
                    sacc[t] = __builtin_amdgcn_wmma_f32_16x16x32_bf16(
                        false, a.v, false, bfr.v, (short)0, sacc[t], false, false);
                }
            }
        }

        // Write scaled scores to LDS
        for (int t = 0; t < 2; ++t) {
            const int tile = tile0 + t;
            const int mt = tile >> 2, nt = tile & 3;
            for (int vv = 0; vv < 8; ++vv)
                Sc[(mt * 16 + vv + lh * 8) * BKV + nt * 16 + l16] = sacc[t][vv] * scale;
        }
        __syncthreads();

        // ---- online softmax, one thread per q row ----
        if (tid < BQ) {
            const int r  = tid;
            const int qg = qBase + r;
            float* srow = &Sc[r * BKV];
            if (mask) {
                for (int c = 0; c < BKV; ++c)
                    if (j * BKV + c > qg) srow[c] = -INFINITY;
            }
            float mold = mrow[r];
            float rm = -INFINITY;
            for (int c = 0; c < BKV; ++c) rm = fmaxf(rm, srow[c]);
            float mnew = fmaxf(mold, rm);
            float corr = __expf(mold - mnew);
            float lsum = 0.f;
            for (int c = 0; c < BKV; ++c) {
                float p = __expf(srow[c] - mnew);
                lsum += p;
                Ps[r * BKV + c] = (bf16)p;
            }
            mrow[r] = mnew;
            lrow[r] = lrow[r] * corr + lsum;
            crow[r] = corr;
        }
        __syncthreads();

        // Rescale output accumulators by row correction
        for (int mt = 0; mt < 4; ++mt)
            for (int vv = 0; vv < 8; ++vv) {
                const float c = crow[mt * 16 + vv + lh * 8];
                for (int nt = 0; nt < 4; ++nt)
                    oacc[mt][nt][vv] = oacc[mt][nt][vv] * c;
            }

        // Load this wave's V chunk, transposed: Vt[h_local][kv]
        {
            bf16* vt = Vt + wave * (HC * BKV);
            for (int i = 0; i < 16; ++i) {
                int f  = i * 32 + lane;   // uint4 index over 512
                int kv = f >> 3;
                int c8 = f & 7;
                uint4 raw = *(const uint4*)(Vg + baseKV + (size_t)kv * H_ + wave * HC + c8 * 8);
                const bf16* e = (const bf16*)&raw;
                for (int x = 0; x < 8; ++x)
                    vt[(c8 * 8 + x) * BKV + kv] = e[x];
            }
        }

        // ---- O += P V for this wave's 64-wide H slice ----
        const bf16* vtb = Vt + wave * (HC * BKV);
        for (int mt = 0; mt < 4; ++mt) {
            for (int ks = 0; ks < 2; ++ks) {
                FragBF a;
                const bf16* ar = &Ps[(mt * 16 + l16) * BKV + ks * 32];
                a.q[0] = *(const uint4*)(ar + lh * 8);
                a.q[1] = *(const uint4*)(ar + 16 + lh * 8);
                for (int nt = 0; nt < 4; ++nt) {
                    FragBF bfr;
                    const bf16* br = &vtb[(nt * 16 + l16) * BKV + ks * 32];
                    bfr.q[0] = *(const uint4*)(br + lh * 8);
                    bfr.q[1] = *(const uint4*)(br + 16 + lh * 8);
                    oacc[mt][nt] = __builtin_amdgcn_wmma_f32_16x16x32_bf16(
                        false, a.v, false, bfr.v, (short)0, oacc[mt][nt], false, false);
                }
            }
        }
    }

    __syncthreads();
    // Epilogue: divide by l, write f32 output
    for (int mt = 0; mt < 4; ++mt)
        for (int vv = 0; vv < 8; ++vv) {
            const int r = mt * 16 + vv + lh * 8;
            const float inv = 1.0f / lrow[r];
            for (int nt = 0; nt < 4; ++nt) {
                const int col = wave * HC + nt * 16 + l16;
                Out[baseQ + (size_t)r * H_ + col] = oacc[mt][nt][vv] * inv;
            }
        }
}

// ---------------------------------------------------------------------------
extern "C" void kernel_launch(void* const* d_in, const int* in_sizes, int n_in,
                              void* d_out, int out_size, void* d_ws, size_t ws_size,
                              hipStream_t stream)
{
    const float* x1 = (const float*)d_in[0];
    const float* x2 = (const float*)d_in[1];
    const float* Wq = (const float*)d_in[2];
    const float* Wk = (const float*)d_in[3];
    const float* Wv = (const float*)d_in[4];
    const int* should_mask = (const int*)d_in[5];
    float* out = (float*)d_out;

    const size_t nQKV = (size_t)B_ * T_ * H_;
    bf16* qb = (bf16*)d_ws;
    bf16* kb = qb + nQKV;
    bf16* vb = kb + nQKV;

    const dim3 gemm_grid((B_ * T_) / 128, H_ / 64, 1);
    qkv_gemm_kernel<<<gemm_grid, 256, 0, stream>>>(x1, Wq, qb);
    qkv_gemm_kernel<<<gemm_grid, 256, 0, stream>>>(x2, Wk, kb);
    qkv_gemm_kernel<<<gemm_grid, 256, 0, stream>>>(x2, Wv, vb);

    (void)hipFuncSetAttribute((const void*)flash_attn_kernel,
                              hipFuncAttributeMaxDynamicSharedMemorySize,
                              ATTN_LDS_BYTES);
    flash_attn_kernel<<<dim3(T_ / BQ, B_, 1), 256, ATTN_LDS_BYTES, stream>>>(
        qb, kb, vb, out, should_mask);
}
